// MLPVolatilityModelImproved_74113955660470
// MI455X (gfx1250) — matrix operand order repaired and checked by reading, not compiled, not run on hardware
//
#include <hip/hip_runtime.h>

// CDNA5 / gfx1250, wave32. D = A*B via v_wmma_f32_16x16x32_bf16.
typedef __attribute__((ext_vector_type(16))) __bf16 v16bf;
typedef __attribute__((ext_vector_type(8)))  float  v8f;

static __device__ __forceinline__ unsigned short f2bf(float f) {
    // round-to-nearest-even f32 -> bf16
    unsigned int u = __float_as_uint(f);
    unsigned int r = (u + 0x7FFFu + ((u >> 16) & 1u)) >> 16;
    return (unsigned short)r;
}

static __device__ __forceinline__ float softplusf(float x) {
    return fmaxf(x, 0.f) + log1pf(expf(-fabsf(x)));
}

// ---------------------------------------------------------------------------
// Kernel 1: residuals u2, bf16 copies for WMMA, deterministic block partials.
// ---------------------------------------------------------------------------
__global__ void prep_kernel(const float* __restrict__ r, const float* __restrict__ volume,
                            const float* __restrict__ a0p, const float* __restrict__ a1p,
                            float* __restrict__ u2f,
                            unsigned short* __restrict__ u2b, unsigned short* __restrict__ volb,
                            float* __restrict__ psum, float* __restrict__ psq,
                            int n, int q) {
    __shared__ float ssum[256];
    __shared__ float ssq[256];
    const int t = blockIdx.x * blockDim.x + threadIdx.x;
    const float a0 = a0p[0], a1 = a1p[0];
    float u2 = 0.f;
    if (t < n) {
        float u = r[t + 1] - a0 - a1 * r[t];
        u2 = u * u;
        u2f[t]      = u2;
        u2b[q + t]  = f2bf(u2);
        volb[q + t] = f2bf(volume[t]);
    }
    ssum[threadIdx.x] = u2;
    ssq[threadIdx.x]  = u2 * u2;
    __syncthreads();
    for (int s = 128; s > 0; s >>= 1) {
        if ((int)threadIdx.x < s) {
            ssum[threadIdx.x] += ssum[threadIdx.x + s];
            ssq[threadIdx.x]  += ssq[threadIdx.x + s];
        }
        __syncthreads();
    }
    if (threadIdx.x == 0) { psum[blockIdx.x] = ssum[0]; psq[blockIdx.x] = ssq[0]; }
}

// ---------------------------------------------------------------------------
// Kernel 2: finish variance reduction, init windows, softplus params,
// build combined reversed weight matrix Bc [2q x k] in bf16.
// ---------------------------------------------------------------------------
__global__ void finalize_kernel(const float* __restrict__ volume,
                                const float* __restrict__ raw_beta0,
                                const float* __restrict__ raw_beta,
                                const float* __restrict__ raw_w0,
                                const float* __restrict__ raw_w,
                                const float* __restrict__ raw_gamma,
                                const float* __restrict__ raw_rho,
                                const float* __restrict__ psum, const float* __restrict__ psq,
                                int nblocks, int n, int q, int k,
                                unsigned short* __restrict__ u2b,
                                unsigned short* __restrict__ volb,
                                unsigned short* __restrict__ Bc,
                                float* __restrict__ pvec, float* __restrict__ pscal) {
    __shared__ float ssum[256];
    __shared__ float ssq[256];
    __shared__ float svar;
    float ls = 0.f, lq = 0.f;
    for (int i = threadIdx.x; i < nblocks; i += 256) { ls += psum[i]; lq += psq[i]; }
    ssum[threadIdx.x] = ls; ssq[threadIdx.x] = lq;
    __syncthreads();
    for (int s = 128; s > 0; s >>= 1) {
        if ((int)threadIdx.x < s) {
            ssum[threadIdx.x] += ssum[threadIdx.x + s];
            ssq[threadIdx.x]  += ssq[threadIdx.x + s];
        }
        __syncthreads();
    }
    if (threadIdx.x == 0) {
        double sum = (double)ssum[0], sq = (double)ssq[0];
        double mean = sum / (double)n;
        double var = (sq - sum * mean) / (double)(n - 1);   // ddof=1
        svar = (float)var;
        pscal[0] = (float)var;
        pscal[1] = softplusf(raw_beta0[0]);
    }
    __syncthreads();
    const float var = svar;
    for (int j = threadIdx.x; j < q; j += 256) {
        u2b[j]  = f2bf(var);
        volb[j] = f2bf(volume[0]);
    }
    for (int i = threadIdx.x; i < k; i += 256) {
        pvec[i]         = softplusf(raw_beta[i]);   // beta
        pvec[k + i]     = softplusf(raw_w0[i]);     // w0
        pvec[2 * k + i] = softplusf(raw_rho[i]);    // rho
    }
    // Bc[jrow, i]: rows 0..q-1 = w[i, q-1-jrow]; rows q..2q-1 = gamma[i, q-1-(jrow-q)]
    for (int idx = threadIdx.x; idx < 2 * q * k; idx += 256) {
        int jrow = idx / k;
        int i = idx - jrow * k;
        float v = (jrow < q) ? softplusf(raw_w[i * q + (q - 1 - jrow)])
                             : softplusf(raw_gamma[i * q + (q - 1 - (jrow - q))]);
        Bc[idx] = f2bf(v);
    }
}

// ---------------------------------------------------------------------------
// Kernel 3: single-wave fused WMMA + sequential GARCH scan.
// Ping-pong double buffering (no register copies), per-lane base pointers so
// all window loads are base + immediate offset, unguarded hot path with a
// separate guarded tail chunk.
// ---------------------------------------------------------------------------
__global__ void __launch_bounds__(32)
scan_kernel(const unsigned short* __restrict__ u2b, const unsigned short* __restrict__ volb,
            const unsigned short* __restrict__ Bc, const float* __restrict__ u2f,
            const float* __restrict__ pvec, const float* __restrict__ pscal,
            float* __restrict__ out, int n) {
    const int lane = threadIdx.x;
    const int m = lane & 15;
    const int h = lane >> 4;
    const float beta  = pvec[lane];
    const float w0    = pvec[32 + lane];
    const float rho   = pvec[64 + lane];
    const float var   = pscal[0];
    const float beta0 = pscal[1];

    union Frag { v16bf v; unsigned short s[16]; };

    // B fragments resident in VGPRs for the whole scan: 4 K-slices x 2 N-tiles.
    // 16-bit B 32x16 layout: element s of lane half h holds K = 8h+s (s<8), 16+8h+(s-8) (s>=8).
    v16bf bfrag[4][2];
#pragma unroll
    for (int ks = 0; ks < 4; ++ks) {
#pragma unroll
        for (int nt = 0; nt < 2; ++nt) {
            Frag f;
#pragma unroll
            for (int idx = 0; idx < 16; ++idx) {
                int K = ks * 32 + h * 8 + (idx & 7) + ((idx >> 3) << 4);
                f.s[idx] = Bc[K * 32 + nt * 16 + m];
            }
            bfrag[ks][nt] = f.v;
        }
    }

    __shared__ float tile[16 * 32];

    // Per-lane base pointers with the lane-dependent (m + h*8) offset folded
    // in: every element address in the hot loop is base + t0 + constant, so
    // the whole A fetch becomes 8 global_load_b128 with immediate offsets.
    const unsigned short* baseU = u2b + (m + h * 8);
    const unsigned short* baseV = volb + (m + h * 8) - 64;

    // 16-bit A 16x32 layout: lane m = M; element (vg,e):
    // K = (vg>>2)*16 + h*8 + (vg&3)*2 + e.  A[t,kk] = u2_full[t+kk] (kk<64)
    // else vol_full[t+kk-64]; the kk<64 select is compile-time per fragment.
    auto loadA = [&](int t0, v16bf (&af)[4]) {
        const unsigned short* pu = baseU + t0;
        const unsigned short* pv = baseV + t0;
#pragma unroll
        for (int ks = 0; ks < 4; ++ks) {
            Frag f;
#pragma unroll
            for (int idx = 0; idx < 16; ++idx) {
                int vg = idx >> 1, e = idx & 1;
                int kk = ks * 32 + ((vg >> 2) << 4) + ((vg & 3) << 1) + e;
                f.s[idx] = (kk < 64) ? pu[kk] : pv[kk];
            }
            af[ks] = f.v;
        }
    };

    float s = var + 1e-6f;
    double accLog = 0.0, accRatio = 0.0;

    // Full (unguarded) chunk: WMMA, prefetch next, redistribute, 16 steps.
    auto bodyFull = [&](int t0, const v16bf (&cur)[4], v16bf (&nxt)[4]) {
        v8f c0 = {}; v8f c1 = {};
#pragma unroll
        for (int ks = 0; ks < 4; ++ks)
            c0 = __builtin_amdgcn_wmma_f32_16x16x32_bf16(false, cur[ks], false, bfrag[ks][0],
                                                         (short)0, c0, false, false);
#pragma unroll
        for (int ks = 0; ks < 4; ++ks)
            c1 = __builtin_amdgcn_wmma_f32_16x16x32_bf16(false, cur[ks], false, bfrag[ks][1],
                                                         (short)0, c1, false, false);

        loadA(t0 + 16, nxt);   // hidden behind the 16 serial steps below

        __syncthreads();
        // f32 D layout: VGPR r holds M = r + 8*h, N = lane%16.
#pragma unroll
        for (int rr = 0; rr < 8; ++rr) {
            int row = rr + 8 * h;
            tile[row * 32 + m]      = c0[rr];
            tile[row * 32 + 16 + m] = c1[rr];
        }
        __syncthreads();

        float sval[16];
#pragma unroll
        for (int j = 0; j < 16; ++j)
            sval[j] = tile[j * 32 + lane] + w0;

        float u2v = u2f[t0 + m];   // lane j holds u2[t0+j] (j<16); halves duplicate

#pragma unroll
        for (int j = 0; j < 16; ++j) {
            float hv = fmaxf(fmaf(rho, s, sval[j]), 0.f);
            float contrib = beta * hv;
#pragma unroll
            for (int off = 16; off > 0; off >>= 1)
                contrib += __shfl_xor(contrib, off, 32);
            float snew = beta0 + contrib + 1e-8f;
            accLog   += (double)logf(snew);
            accRatio += (double)(__shfl(u2v, j, 32) / snew);
            s = snew;
        }
    };

    // Guarded tail chunk (t >= n steps masked out).
    auto bodyTail = [&](int t0, const v16bf (&cur)[4]) {
        v8f c0 = {}; v8f c1 = {};
#pragma unroll
        for (int ks = 0; ks < 4; ++ks)
            c0 = __builtin_amdgcn_wmma_f32_16x16x32_bf16(false, cur[ks], false, bfrag[ks][0],
                                                         (short)0, c0, false, false);
#pragma unroll
        for (int ks = 0; ks < 4; ++ks)
            c1 = __builtin_amdgcn_wmma_f32_16x16x32_bf16(false, cur[ks], false, bfrag[ks][1],
                                                         (short)0, c1, false, false);
        __syncthreads();
#pragma unroll
        for (int rr = 0; rr < 8; ++rr) {
            int row = rr + 8 * h;
            tile[row * 32 + m]      = c0[rr];
            tile[row * 32 + 16 + m] = c1[rr];
        }
        __syncthreads();

        float sval[16];
#pragma unroll
        for (int j = 0; j < 16; ++j)
            sval[j] = tile[j * 32 + lane] + w0;

        int tu = t0 + m;
        float u2v = u2f[tu < n ? tu : (n - 1)];

#pragma unroll
        for (int j = 0; j < 16; ++j) {
            float hv = fmaxf(fmaf(rho, s, sval[j]), 0.f);
            float contrib = beta * hv;
#pragma unroll
            for (int off = 16; off > 0; off >>= 1)
                contrib += __shfl_xor(contrib, off, 32);
            float snew = beta0 + contrib + 1e-8f;
            if (t0 + j < n) {
                accLog   += (double)logf(snew);
                accRatio += (double)(__shfl(u2v, j, 32) / snew);
                s = snew;
            }
        }
    };

    const int nFull = n >> 4;
    v16bf aA[4], aB[4];
    loadA(0, aA);

    int c = 0;
    for (; c + 1 < nFull; c += 2) {
        bodyFull(c << 4, aA, aB);
        bodyFull((c + 1) << 4, aB, aA);
    }
    bool curIsA = true;
    if (c < nFull) { bodyFull(c << 4, aA, aB); curIsA = false; }
    if (n & 15) {
        if (curIsA) bodyTail(nFull << 4, aA);
        else        bodyTail(nFull << 4, aB);
    }

    if (lane == 0) {
        double nll = 0.5 * (double)n * 1.8378770664093453  // log(2*pi)
                   + 0.5 * accLog + 0.5 * accRatio;
        out[0] = (float)nll;
    }
}

// ---------------------------------------------------------------------------
extern "C" void kernel_launch(void* const* d_in, const int* in_sizes, int n_in,
                              void* d_out, int out_size, void* d_ws, size_t ws_size,
                              hipStream_t stream) {
    const float* r      = (const float*)d_in[0];
    const float* volume = (const float*)d_in[1];
    const float* a0     = (const float*)d_in[2];
    const float* a1     = (const float*)d_in[3];
    const float* rb0    = (const float*)d_in[4];
    const float* rbeta  = (const float*)d_in[5];
    const float* rw0    = (const float*)d_in[6];
    const float* rw     = (const float*)d_in[7];
    const float* rgamma = (const float*)d_in[8];
    const float* rrho   = (const float*)d_in[9];

    const int T = in_sizes[0];
    const int n = T - 1;
    const int k = in_sizes[5];            // 32
    const int q = in_sizes[7] / k;        // 64

    char* ws = (char*)d_ws;
    size_t off = 0;
    auto carve = [&](size_t bytes) {
        char* p = ws + off;
        off = (off + bytes + 255) & ~(size_t)255;
        return p;
    };
    float*          u2f  = (float*)         carve((size_t)(n + 32) * sizeof(float));
    unsigned short* u2b  = (unsigned short*)carve((size_t)(q + n + 96) * sizeof(unsigned short));
    unsigned short* volb = (unsigned short*)carve((size_t)(q + n + 96) * sizeof(unsigned short));
    unsigned short* Bc   = (unsigned short*)carve((size_t)(2 * q * k) * sizeof(unsigned short));
    float*          pvec = (float*)         carve((size_t)(3 * k) * sizeof(float));
    float*          pscal= (float*)         carve(16 * sizeof(float));
    const int nblocks = (n + 255) / 256;
    float*          psum = (float*)         carve((size_t)nblocks * sizeof(float));
    float*          psq  = (float*)         carve((size_t)nblocks * sizeof(float));
    (void)ws_size; (void)n_in; (void)out_size;

    prep_kernel<<<nblocks, 256, 0, stream>>>(r, volume, a0, a1, u2f, u2b, volb, psum, psq, n, q);
    finalize_kernel<<<1, 256, 0, stream>>>(volume, rb0, rbeta, rw0, rw, rgamma, rrho,
                                           psum, psq, nblocks, n, q, k,
                                           u2b, volb, Bc, pvec, pscal);
    scan_kernel<<<1, 32, 0, stream>>>(u2b, volb, Bc, u2f, pvec, pscal, (float*)d_out, n);
}
